// FlowVN_78718160601601
// MI455X (gfx1250) — matrix operand order repaired and checked by reading, not compiled, not run on hardware
//
#include <hip/hip_runtime.h>

typedef __attribute__((ext_vector_type(16))) _Float16 v16h;
typedef __attribute__((ext_vector_type(8)))  float    v8f;

#define SV   884736L   // V (channel) stride in u_t_1 elements: T*D*H*W*2
#define KACT 31
#define COUT 24

// Build one 32x16 K-chunk B fragment (contiguous k order: element h holds k = KBB + h).
// C (K-chunk) and KBB (lane-half K base: 0 for lanes 0-15, 16 for lanes 16-31)
// are compile-time so the im2col decomposition folds into immediate DS offsets.
template<int C, int KBB>
__device__ __forceinline__ v16h buildBf(const _Float16* __restrict__ sIn, int baseHalf) {
  v16h bf;
#pragma unroll
  for (int h = 0; h < 16; ++h) {
    const int kk = C * 32 + KBB + h;
    if (kk < 108) {
      const int ci = kk / 27, r = kk % 27;
      const int dz = r / 9, r9 = r % 9, dy = r9 / 3, dx = r9 - dy * 3;
      const int off = ((ci * 3 + dz) * 14 + dy) * 50 + dx;   // constant
      bf[h] = sIn[baseHalf + off];                            // ds_load_u16 imm-offset
    } else {
      bf[h] = (_Float16)0.0f;                                 // K padding
    }
  }
  return bf;
}

__global__ __launch_bounds__(256) void flowvn_view_kernel(
    const float* __restrict__ u, const float* __restrict__ kw,
    const float* __restrict__ wc, const float* __restrict__ mu,
    const float* __restrict__ sigp, float* __restrict__ outp,
    int Bn, int n0, int n1, int n2,
    long sB, long s0, long s1, long s2)
{
  __shared__ float    sInF[4 * 3 * 14 * 48];  // raw f32 landing zone (async DMA dest)
  __shared__ _Float16 sIn[4 * 3 * 14 * 50];   // f16 halo-padded gather tile
  __shared__ _Float16 sAw[32 * 128];          // im2col A-matrix, zero padded
  __shared__ float    sCoef[COUT * KACT];
  __shared__ float    sMu[32];

  const int tid  = threadIdx.x;
  const int lane = tid & 31;
  const int wvid = tid >> 5;

  const int C1 = (n1 + 11) / 12;
  int bid = blockIdx.x;
  const int c1 = bid % C1; bid /= C1;
  const int z  = bid % n0;
  const int b  = bid / n0;
  const int r0 = c1 * 12;
  const int n1c = (n1 - r0) < 12 ? (n1 - r0) : 12;

  const long bbase = (long)(b >> 1) * sB + (long)(b & 1);

  // ---- async-DMA the in-bounds f32 interior straight into LDS (ASYNCcnt) ----
  const int nIn = 4 * 3 * 14 * n2;
  for (int idx = tid; idx < nIn; idx += 256) {
    int pp = idx % n2; int rest = idx / n2;
    int j  = rest % 14; int cz  = rest / 14;
    int dz = cz % 3;    int ci  = cz / 3;
    int zz = z + dz - 1, rr = r0 - 1 + j;
    if (zz >= 0 && zz < n0 && rr >= 0 && rr < n1) {
      const float* gp = u + (bbase + (long)ci * SV + (long)zz * s0
                             + (long)rr * s1 + (long)pp * s2);
      unsigned lo = (unsigned)(size_t)&sInF[((ci * 3 + dz) * 14 + j) * 48 + pp];
      unsigned long long ga = (unsigned long long)(size_t)gp;
      asm volatile("global_load_async_to_lds_b32 %0, %1, off"
                   :: "v"(lo), "v"(ga) : "memory");
    }
  }

  // ---- stage conv weights as 32x128 f16 im2col A-matrix (zero padded) ----
  for (int idx = tid; idx < 32 * 128; idx += 256) {
    int m = idx >> 7, kk = idx & 127;
    float v = 0.0f;
    if (m < COUT && kk < 108) {
      int ci = kk / 27, r = kk % 27;   // r = dz*9+dy*3+dx matches OIDHW inner order
      v = kw[(m * 4 + ci) * 27 + r];
    }
    sAw[idx] = (_Float16)v;
  }
  // ---- RBF weights / centers ----
  for (int idx = tid; idx < COUT * KACT; idx += 256) sCoef[idx] = wc[idx];
  if (tid < 32) sMu[tid] = (tid < KACT) ? mu[tid] : 0.0f;

  asm volatile("s_wait_asynccnt 0" ::: "memory");
  __syncthreads();

  // ---- f32 -> f16, add halo + OOB zero padding (one-time LDS->LDS pass) ----
  const int n2p = n2 + 2;
  for (int idx = tid; idx < 4 * 3 * 14 * n2p; idx += 256) {
    int kcol = idx % n2p; int rest = idx / n2p;
    int j  = rest % 14;   int cz  = rest / 14;
    int dz = cz % 3;      int ci  = cz / 3;
    int zz = z + dz - 1, rr = r0 - 1 + j, pp = kcol - 1;
    float v = 0.0f;
    if (zz >= 0 && zz < n0 && rr >= 0 && rr < n1 && pp >= 0 && pp < n2)
      v = sInF[((ci * 3 + dz) * 14 + j) * 48 + pp];
    sIn[((ci * 3 + dz) * 14 + j) * 50 + kcol] = (_Float16)v;
  }
  __syncthreads();

  const float sg  = sigp[0];
  const float inv = 0.5f / (sg * sg);

  // ---- per-lane A fragments, contiguous k order (element h holds k = 16*(lane/16) + h), same as B ----
  v16h afrag[2][4];
  {
    const int arow = lane & 15;
    const int kba  = (lane >= 16) ? 16 : 0;
#pragma unroll
    for (int mt = 0; mt < 2; ++mt)
#pragma unroll
      for (int c = 0; c < 4; ++c)
#pragma unroll
        for (int h = 0; h < 16; ++h) {
          int kk = c * 32 + kba + h;
          afrag[mt][c][h] = sAw[(arow + mt * 16) * 128 + kk];
        }
  }

  const int nPos   = n1c * n2;
  const int nTiles = (nPos + 15) >> 4;
  const int col  = lane & 15;
  const int rsel = (lane >= 16) ? 8 : 0;

  for (int tile = wvid; tile < nTiles; tile += 8) {
    int pos  = tile * 16 + col;
    int posc = (pos < nPos) ? pos : (nPos - 1);
    int p1 = posc / n2;
    int p2 = posc - p1 * n2;
    int baseHalf = p1 * 50 + p2;

    v16h bf0, bf1, bf2, bf3;
    if (lane < 16) {                      // KBB folds to 0: imm-offset gathers
      bf0 = buildBf<0, 0>(sIn, baseHalf);
      bf1 = buildBf<1, 0>(sIn, baseHalf);
      bf2 = buildBf<2, 0>(sIn, baseHalf);
      bf3 = buildBf<3, 0>(sIn, baseHalf);
    } else {                              // KBB folds to 16
      bf0 = buildBf<0, 16>(sIn, baseHalf);
      bf1 = buildBf<1, 16>(sIn, baseHalf);
      bf2 = buildBf<2, 16>(sIn, baseHalf);
      bf3 = buildBf<3, 16>(sIn, baseHalf);
    }

    v8f acc0 = {}; v8f acc1 = {};
    acc0 = __builtin_amdgcn_wmma_f32_16x16x32_f16(false, afrag[0][0], false, bf0, (short)0, acc0, false, false);
    acc1 = __builtin_amdgcn_wmma_f32_16x16x32_f16(false, afrag[1][0], false, bf0, (short)0, acc1, false, false);
    acc0 = __builtin_amdgcn_wmma_f32_16x16x32_f16(false, afrag[0][1], false, bf1, (short)0, acc0, false, false);
    acc1 = __builtin_amdgcn_wmma_f32_16x16x32_f16(false, afrag[1][1], false, bf1, (short)0, acc1, false, false);
    acc0 = __builtin_amdgcn_wmma_f32_16x16x32_f16(false, afrag[0][2], false, bf2, (short)0, acc0, false, false);
    acc1 = __builtin_amdgcn_wmma_f32_16x16x32_f16(false, afrag[1][2], false, bf2, (short)0, acc1, false, false);
    acc0 = __builtin_amdgcn_wmma_f32_16x16x32_f16(false, afrag[0][3], false, bf3, (short)0, acc0, false, false);
    acc1 = __builtin_amdgcn_wmma_f32_16x16x32_f16(false, afrag[1][3], false, bf3, (short)0, acc1, false, false);

    const bool colOK = (pos < nPos);
#pragma unroll
    for (int mt = 0; mt < 2; ++mt)
#pragma unroll
      for (int i = 0; i < 8; ++i) {
        int crow = mt * 16 + i + rsel;             // D layout: VGPR i -> M=i / i+8
        if (crow < COUT && colOK) {
          float x = (mt == 0) ? acc0[i] : acc1[i];
          float s = 0.0f;
#pragma unroll
          for (int j = 0; j < KACT; ++j) {
            float d = x - sMu[j];
            s += sCoef[crow * KACT + j] * __expf(-(d * d) * inv);
          }
          long oidx = (((long)b * COUT + crow) * n0 + z) * (long)n1 * n2
                    + (long)(r0 + p1) * n2 + p2;
          outp[oidx] = s;
        }
      }
  }
}

extern "C" void kernel_launch(void* const* d_in, const int* in_sizes, int n_in,
                              void* d_out, int out_size, void* d_ws, size_t ws_size,
                              hipStream_t stream) {
  const float* u     = (const float*)d_in[0];
  const float* k_xyz = (const float*)d_in[3];
  const float* k_xyt = (const float*)d_in[4];
  const float* k_yzt = (const float*)d_in[5];
  const float* k_xzt = (const float*)d_in[6];
  const float* w1    = (const float*)d_in[7];
  const float* w2    = (const float*)d_in[8];
  const float* w3    = (const float*)d_in[9];
  const float* w4    = (const float*)d_in[10];
  const float* mu    = (const float*)d_in[11];
  const float* sig   = (const float*)d_in[12];
  float* out = (float*)d_out;

  const long VIEW = 24L * 24 * 16 * 48 * 48;       // 21,233,664 per view
  const long sT = 73728, sD = 4608, sH = 96, sW = 2;

  // view xyz: batch=(t,r) stride sT; spatial (D,H,W)
  flowvn_view_kernel<<<24 * 16 * 4, 256, 0, stream>>>(u, k_xyz, w1, mu, sig, out,
      24, 16, 48, 48, sT, sD, sH, sW);
  // view xyt: batch=(w,r) stride sW; spatial (D,H,T)
  flowvn_view_kernel<<<96 * 16 * 4, 256, 0, stream>>>(u, k_xyt, w2, mu, sig, out + VIEW,
      96, 16, 48, 12, sW, sD, sH, sT);
  // view yzt: batch=(d,r) stride sD; spatial (H,W,T)
  flowvn_view_kernel<<<32 * 48 * 4, 256, 0, stream>>>(u, k_yzt, w3, mu, sig, out + 2 * VIEW,
      32, 48, 48, 12, sD, sH, sW, sT);
  // view xzt: batch=(h,r) stride sH; spatial (D,W,T)
  flowvn_view_kernel<<<96 * 16 * 4, 256, 0, stream>>>(u, k_xzt, w4, mu, sig, out + 3 * VIEW,
      96, 16, 48, 12, sH, sD, sW, sT);
}
